// Gating_47785806135840
// MI455X (gfx1250) — matrix operand order repaired and checked by reading, not compiled, not run on hardware
//
#include <hip/hip_runtime.h>
#include <cmath>

#define Bsz   8192
#define Dsz   1024
#define Esz   16
#define Hsz   1024
#define TOPK  8

#define BM  128
#define BN  128
#define BK  64
#define LDK 88   // halfword stride: 176B rows -> 16B aligned, conflict-free banks
#define NSTAGES ((Esz * Dsz) / BK)   // 256
#define BUFH ((BM + BN) * LDK)       // halfwords per LDS buffer
#define BUFBYTES (BUFH * 2)

typedef __attribute__((ext_vector_type(16))) __bf16 bf16x16;
typedef __attribute__((ext_vector_type(8)))  __bf16 bf16x8;
typedef __attribute__((ext_vector_type(8)))  float  f32x8;

// ---------------------------------------------------------------------------
// Kernel 1: gating network (wave per row; scalar tail per row; LDS-reduced
// importance/load stats -> 16 global atomics per block).
// ---------------------------------------------------------------------------
__global__ __launch_bounds__(256) void gating_kernel(
    const float* __restrict__ x, const float* __restrict__ wg,
    const float* __restrict__ wn, const float* __restrict__ noise,
    float* __restrict__ gates, float* __restrict__ g_imp,
    float* __restrict__ g_load)
{
    __shared__ float sm[8][32];
    __shared__ float blkImp[16];
    __shared__ float blkLoad[16];

    const int tid = threadIdx.x;
    if (tid < 16) { blkImp[tid] = 0.f; blkLoad[tid] = 0.f; }

    const int wid  = tid >> 5;
    const int lane = tid & 31;
    const int b    = blockIdx.x * 8 + wid;
    const int e    = lane & 15;
    const float* w  = (lane < 16) ? wg : wn;
    const float* xr = x + (size_t)b * Dsz;

    float acc = 0.f;
    for (int d = 0; d < Dsz; d += 4) {
        const float4 xv = *(const float4*)(xr + d);
        acc += xv.x * w[(d + 0) * Esz + e];
        acc += xv.y * w[(d + 1) * Esz + e];
        acc += xv.z * w[(d + 2) * Esz + e];
        acc += xv.w * w[(d + 3) * Esz + e];
    }
    sm[wid][lane] = acc;
    __syncthreads();

    if (tid < 8) {
        const int r  = tid;
        const int bb = blockIdx.x * 8 + r;
        float clean[16], stddev[16], noisy[16], probs[16];
        float mx = -1e30f;
        for (int i = 0; i < 16; ++i) {
            clean[i] = sm[r][i];
            const float raw = sm[r][16 + i];
            const float sp  = (raw > 0.f) ? (raw + log1pf(expf(-raw)))
                                          : log1pf(expf(raw));
            stddev[i] = sp + 0.01f;                       // NOISE_EPS
            noisy[i]  = clean[i] + noise[bb * Esz + i] * stddev[i];
            mx = fmaxf(mx, noisy[i]);
        }
        float s = 0.f;
        for (int i = 0; i < 16; ++i) { probs[i] = expf(noisy[i] - mx); s += probs[i]; }
        const float invs = 1.f / s;
        for (int i = 0; i < 16; ++i) probs[i] *= invs;

        float tv[9]; int ti[9]; bool used[16];
        for (int i = 0; i < 16; ++i) used[i] = false;
        for (int j = 0; j < 9; ++j) {
            float bv = -1.f; int bi = 0;
            for (int i = 0; i < 16; ++i)
                if (!used[i] && probs[i] > bv) { bv = probs[i]; bi = i; }
            used[bi] = true; tv[j] = bv; ti[j] = bi;
        }
        float ssum = 0.f;
        for (int j = 0; j < TOPK; ++j) ssum += tv[j];
        const float rs = 1.f / (ssum + 1e-6f);
        float g[16];
        for (int i = 0; i < 16; ++i) g[i] = 0.f;
        for (int j = 0; j < TOPK; ++j) g[ti[j]] = tv[j] * rs;

        const float thr_in = tv[TOPK], thr_out = tv[TOPK - 1];
        for (int i = 0; i < 16; ++i) {
            gates[(size_t)bb * Esz + i] = g[i];
            const bool  in  = noisy[i] > thr_in;
            const float thr = in ? thr_in : thr_out;
            const float z   = (clean[i] - thr) / stddev[i];
            const float p   = 0.5f * (1.f + erff(z * 0.70710678118654752f));
            atomicAdd(&blkLoad[i], p);
            atomicAdd(&blkImp[i], g[i]);
        }
    }
    __syncthreads();
    if (tid < 16) {
        atomicAdd(&g_imp[tid],  blkImp[tid]);
        atomicAdd(&g_load[tid], blkLoad[tid]);
    }
}

// ---------------------------------------------------------------------------
// Kernel 2: loss = 0.01 * (cv^2(importance) + cv^2(load)), ddof=1.
// ---------------------------------------------------------------------------
__global__ void loss_kernel(const float* __restrict__ g_imp,
                            const float* __restrict__ g_load,
                            float* __restrict__ out_loss)
{
    if (threadIdx.x == 0 && blockIdx.x == 0) {
        float m1 = 0.f, m2 = 0.f;
        for (int i = 0; i < 16; ++i) { m1 += g_imp[i]; m2 += g_load[i]; }
        m1 *= (1.f / 16.f); m2 *= (1.f / 16.f);
        float v1 = 0.f, v2 = 0.f;
        for (int i = 0; i < 16; ++i) {
            const float d1 = g_imp[i] - m1, d2 = g_load[i] - m2;
            v1 += d1 * d1; v2 += d2 * d2;
        }
        v1 *= (1.f / 15.f); v2 *= (1.f / 15.f);
        *out_loss = 0.01f * (v1 / (m1 * m1 + 1e-10f) + v2 / (m2 * m2 + 1e-10f));
    }
}

// ---------------------------------------------------------------------------
// Prep kernels: one-time f32 -> bf16 conversion of x (row-major) and
// expert_w transposed to [e][h][d] so both GEMM operands stream as
// contiguous bf16 rows for GLOBAL_LOAD_ASYNC_TO_LDS_B128.
// ---------------------------------------------------------------------------
__global__ __launch_bounds__(256) void convert_x_kernel(
    const float* __restrict__ x, __bf16* __restrict__ xb)
{
    const size_t i = ((size_t)blockIdx.x * 256 + threadIdx.x) * 4;
    const float4 v = *(const float4*)(x + i);
    __bf16* d = xb + i;
    d[0] = (__bf16)v.x; d[1] = (__bf16)v.y;
    d[2] = (__bf16)v.z; d[3] = (__bf16)v.w;
}

__global__ __launch_bounds__(256) void transpose_w_kernel(
    const float* __restrict__ ew, __bf16* __restrict__ wbT)
{
    __shared__ float t[32][33];
    const int e  = blockIdx.z;
    const int hb = blockIdx.x * 32;
    const int db = blockIdx.y * 32;
    const int c  = threadIdx.x & 31;
    const int r0 = threadIdx.x >> 5;
    #pragma unroll
    for (int i = 0; i < 4; ++i) {
        const int r = r0 + i * 8;
        t[r][c] = ew[((size_t)e * Dsz + (db + r)) * Hsz + hb + c];
    }
    __syncthreads();
    #pragma unroll
    for (int i = 0; i < 4; ++i) {
        const int r = r0 + i * 8;
        wbT[((size_t)e * Hsz + (hb + r)) * Dsz + db + c] = (__bf16)t[c][r];
    }
}

// ---------------------------------------------------------------------------
// Fragment helpers (16-bit operand layouts, CDNA5 ISA):
//  A 16x32: lane = row; lanes<16 hold K 0-7 & 16-23, lanes>=16 hold 8-15 & 24-31
//  B 32x16: lane = col; lanes<16 hold K 0-7 & 16-23, lanes>=16 hold 8-15 & 24-31
// ---------------------------------------------------------------------------
__device__ __forceinline__ bf16x16 frag_a(const __bf16* p) {
    const bf16x8 lo = *(const bf16x8*)p;
    const bf16x8 hi = *(const bf16x8*)(p + 16);
    return __builtin_shufflevector(lo, hi, 0, 1, 2, 3, 4, 5, 6, 7,
                                           8, 9, 10, 11, 12, 13, 14, 15);
}
__device__ __forceinline__ bf16x16 frag_b(const __bf16* p) {
    const bf16x8 lo = *(const bf16x8*)p;
    const bf16x8 hi = *(const bf16x8*)(p + 16);
    return __builtin_shufflevector(lo, hi, 0, 1, 2, 3, 4, 5, 6, 7,
                                           8, 9, 10, 11, 12, 13, 14, 15);
}

// ---------------------------------------------------------------------------
// Kernel 3a (preferred): single dense GEMM, M=8192 N=1024 K=16384,
// A'[b,e*D+d] = gates[b,e]*x[b,d] with the gate applied as a packed-bf16
// fragment scale (one row per lane). Tiles staged via
// GLOBAL_LOAD_ASYNC_TO_LDS_B128 into double-buffered LDS; stage s+1 DMA
// overlaps stage s WMMA, synchronized with s_wait_asynccnt.
// ---------------------------------------------------------------------------
__global__ __launch_bounds__(256) void moe_gemm_async(
    const __bf16* __restrict__ xb, const __bf16* __restrict__ wbT,
    const float* __restrict__ expert_b, const float* __restrict__ gates,
    float* __restrict__ y)
{
    __shared__ __attribute__((aligned(128))) __bf16 smem[2 * BUFH];
    __shared__ float gs[BM * Esz];

    const int tid = threadIdx.x;
    const int m0  = blockIdx.y * BM;
    const int n0  = blockIdx.x * BN;

    for (int i = tid; i < BM * Esz; i += 256)
        gs[i] = gates[(size_t)(m0 + (i >> 4)) * Esz + (i & 15)];

    const int wid   = tid >> 5;
    const int lane  = tid & 31;
    const int wm    = wid >> 2;
    const int wn    = wid & 3;
    const int lrow  = lane & 15;
    const int khalf = (lane >> 4) << 3;    // A operand half-select (0/8)
    const int kbB16 = (lane >> 4) << 3;    // B operand half-select (0/8)

    // Per-thread async copy slots: 4 A chunks + 4 B chunks of 16B per stage.
    unsigned ldsA0[4], ldsB0[4], aElem[4], bElem[4];
    #pragma unroll
    for (int i = 0; i < 4; ++i) {
        const int c   = tid + i * 256;       // 0..1023
        const int row = c >> 3;              // 0..127
        const int kc  = (c & 7) * 8;         // 0..56
        aElem[i] = (unsigned)((m0 + row) * Dsz + kc);
        bElem[i] = (unsigned)((n0 + row) * Dsz + kc);
        ldsA0[i] = (unsigned)(uintptr_t)&smem[row * LDK + kc];
        ldsB0[i] = (unsigned)(uintptr_t)&smem[BM * LDK + row * LDK + kc];
    }

    f32x8 acc[4][2] = {};
    __bf16 gsc[4];

    auto issue_stage = [&](int kk, int buf) {
        const unsigned add   = buf ? (unsigned)BUFBYTES : 0u;
        const unsigned dbase = (unsigned)(kk & (Dsz - 1));
        const unsigned ek    = ((unsigned)(kk >> 10) << 20) + dbase;
        #pragma unroll
        for (int i = 0; i < 4; ++i)
            asm volatile("global_load_async_to_lds_b128 %0, %1, %2 offset:0"
                         :: "v"(ldsA0[i] + add), "v"((aElem[i] + dbase) * 2u),
                            "s"(xb) : "memory");
        #pragma unroll
        for (int i = 0; i < 4; ++i)
            asm volatile("global_load_async_to_lds_b128 %0, %1, %2 offset:0"
                         :: "v"(ldsB0[i] + add), "v"((bElem[i] + ek) * 2u),
                            "s"(wbT) : "memory");
    };

    issue_stage(0, 0);
    __syncthreads();   // gs visible before first compute reads it

    for (int s = 0; s < NSTAGES; ++s) {
        const int kk = s * BK;
        if ((kk & (Dsz - 1)) == 0) {       // expert boundary: refresh gates
            const int e = kk >> 10;
            #pragma unroll
            for (int mi = 0; mi < 4; ++mi)
                gsc[mi] = (__bf16)gs[(wm * 64 + mi * 16 + lrow) * Esz + e];
        }
        if (s + 1 < NSTAGES) {
            issue_stage(kk + BK, (s + 1) & 1);
            asm volatile("s_wait_asynccnt 0x8" ::: "memory");  // stage s done
        } else {
            asm volatile("s_wait_asynccnt 0x0" ::: "memory");
        }
        __syncthreads();

        const __bf16* Ab = smem + (s & 1) * BUFH;
        const __bf16* Bb = Ab + BM * LDK;
        #pragma unroll
        for (int ks = 0; ks < 2; ++ks) {
            bf16x16 afrag[4], bfrag[2];
            #pragma unroll
            for (int mi = 0; mi < 4; ++mi)
                afrag[mi] = frag_a(&Ab[(wm * 64 + mi * 16 + lrow) * LDK + ks * 32 + khalf]);
            #pragma unroll
            for (int ni = 0; ni < 2; ++ni)
                bfrag[ni] = frag_b(&Bb[(wn * 32 + ni * 16 + lrow) * LDK + ks * 32 + kbB16]);
            #pragma unroll
            for (int mi = 0; mi < 4; ++mi) {
                const bf16x16 as = afrag[mi] * gsc[mi];   // v_pk_mul_bf16 x8
                #pragma unroll
                for (int ni = 0; ni < 2; ++ni)
                    acc[mi][ni] = __builtin_amdgcn_wmma_f32_16x16x32_bf16(
                        false, as, false, bfrag[ni],
                        (short)0, acc[mi][ni], false, false);
            }
        }
        __syncthreads();   // all reads of this buffer done before refill
    }

    const int roff = (lane >> 4) << 3;
    #pragma unroll
    for (int ni = 0; ni < 2; ++ni) {
        const int n_g = n0 + wn * 32 + ni * 16 + lrow;
        float eb[16];
        #pragma unroll
        for (int e2 = 0; e2 < 16; ++e2) eb[e2] = expert_b[e2 * Hsz + n_g];
        #pragma unroll
        for (int mi = 0; mi < 4; ++mi) {
            #pragma unroll
            for (int r = 0; r < 8; ++r) {
                const int row_l = wm * 64 + mi * 16 + roff + r;
                float bias = 0.f;
                #pragma unroll
                for (int e2 = 0; e2 < 16; ++e2)
                    bias += gs[row_l * Esz + e2] * eb[e2];
                y[(size_t)(m0 + row_l) * Hsz + n_g] = acc[mi][ni][r] + bias;
            }
        }
    }
}

// ---------------------------------------------------------------------------
// Kernel 3b (fallback if ws too small for bf16 staging): fused convert GEMM.
// ---------------------------------------------------------------------------
__global__ __launch_bounds__(256) void moe_gemm_fallback(
    const float* __restrict__ x, const float* __restrict__ expert_w,
    const float* __restrict__ expert_b, const float* __restrict__ gates,
    float* __restrict__ y)
{
    __shared__ __bf16 As[BM * LDK];
    __shared__ __bf16 Bs[BN * LDK];
    __shared__ float  gs[BM * Esz];

    const int tid = threadIdx.x;
    const int m0  = blockIdx.y * BM;
    const int n0  = blockIdx.x * BN;

    for (int i = tid; i < BM * Esz; i += 256)
        gs[i] = gates[(size_t)(m0 + (i >> 4)) * Esz + (i & 15)];

    const int wid   = tid >> 5;
    const int lane  = tid & 31;
    const int wm    = wid >> 2;
    const int wn    = wid & 3;
    const int lrow  = lane & 15;
    const int khalf = (lane >> 4) << 3;
    const int kbB16 = (lane >> 4) << 3;

    f32x8 acc[4][2] = {};
    __syncthreads();

    for (int kk = 0; kk < Esz * Dsz; kk += BK) {
        const int e     = kk >> 10;
        const int dbase = kk & (Dsz - 1);
        #pragma unroll
        for (int i = 0; i < 8; ++i) {
            const int idx = i * 256 + tid;
            const int row = idx >> 4;
            const int q   = idx & 15;
            const float4 v  = *(const float4*)(x + (size_t)(m0 + row) * Dsz + dbase + q * 4);
            const float  gv = gs[row * Esz + e];
            __bf16* dst = &As[row * LDK + q * 4];
            dst[0] = (__bf16)(v.x * gv); dst[1] = (__bf16)(v.y * gv);
            dst[2] = (__bf16)(v.z * gv); dst[3] = (__bf16)(v.w * gv);
        }
        const float* wsrc = expert_w + ((size_t)e * Dsz + dbase) * Hsz + n0;
        #pragma unroll
        for (int i = 0; i < 8; ++i) {
            const int idx = i * 256 + tid;
            const int k   = idx >> 5;
            const int n4  = idx & 31;
            const float4 v = *(const float4*)(wsrc + (size_t)k * Hsz + n4 * 4);
            Bs[(n4 * 4 + 0) * LDK + k] = (__bf16)v.x;
            Bs[(n4 * 4 + 1) * LDK + k] = (__bf16)v.y;
            Bs[(n4 * 4 + 2) * LDK + k] = (__bf16)v.z;
            Bs[(n4 * 4 + 3) * LDK + k] = (__bf16)v.w;
        }
        __syncthreads();
        #pragma unroll
        for (int ks = 0; ks < 2; ++ks) {
            bf16x16 afrag[4], bfrag[2];
            #pragma unroll
            for (int mi = 0; mi < 4; ++mi)
                afrag[mi] = frag_a(&As[(wm * 64 + mi * 16 + lrow) * LDK + ks * 32 + khalf]);
            #pragma unroll
            for (int ni = 0; ni < 2; ++ni)
                bfrag[ni] = frag_b(&Bs[(wn * 32 + ni * 16 + lrow) * LDK + ks * 32 + kbB16]);
            #pragma unroll
            for (int mi = 0; mi < 4; ++mi)
                #pragma unroll
                for (int ni = 0; ni < 2; ++ni)
                    acc[mi][ni] = __builtin_amdgcn_wmma_f32_16x16x32_bf16(
                        false, afrag[mi], false, bfrag[ni],
                        (short)0, acc[mi][ni], false, false);
        }
        __syncthreads();
    }

    const int roff = (lane >> 4) << 3;
    #pragma unroll
    for (int ni = 0; ni < 2; ++ni) {
        const int n_g = n0 + wn * 32 + ni * 16 + lrow;
        float eb[16];
        #pragma unroll
        for (int e2 = 0; e2 < 16; ++e2) eb[e2] = expert_b[e2 * Hsz + n_g];
        #pragma unroll
        for (int mi = 0; mi < 4; ++mi) {
            #pragma unroll
            for (int r = 0; r < 8; ++r) {
                const int row_l = wm * 64 + mi * 16 + roff + r;
                float bias = 0.f;
                #pragma unroll
                for (int e2 = 0; e2 < 16; ++e2)
                    bias += gs[row_l * Esz + e2] * eb[e2];
                y[(size_t)(m0 + row_l) * Hsz + n_g] = acc[mi][ni][r] + bias;
            }
        }
    }
}

// ---------------------------------------------------------------------------
extern "C" void kernel_launch(void* const* d_in, const int* in_sizes, int n_in,
                              void* d_out, int out_size, void* d_ws, size_t ws_size,
                              hipStream_t stream)
{
    (void)in_sizes; (void)n_in; (void)out_size;
    const float* x        = (const float*)d_in[0];
    const float* w_gate   = (const float*)d_in[1];
    const float* w_noise  = (const float*)d_in[2];
    const float* expert_w = (const float*)d_in[3];
    const float* expert_b = (const float*)d_in[4];
    const float* noise    = (const float*)d_in[5];
    float* out = (float*)d_out;

    float* ws     = (float*)d_ws;
    float* g_imp  = ws;                  // 16 floats
    float* g_load = ws + 16;             // 16 floats
    float* gates  = ws + 1024;           // B*E floats @ 4KB offset

    hipMemsetAsync(ws, 0, 32 * sizeof(float), stream);
    gating_kernel<<<Bsz / 8, 256, 0, stream>>>(x, w_gate, w_noise, noise,
                                               gates, g_imp, g_load);
    loss_kernel<<<1, 64, 0, stream>>>(g_imp, g_load, out + (size_t)Bsz * Hsz);

    const size_t XB_OFF  = (size_t)1 << 20;                       // 1 MB
    const size_t XB_BYT  = (size_t)Bsz * Dsz * 2;                 // 16 MB
    const size_t WB_BYT  = (size_t)Esz * Dsz * Hsz * 2;           // 32 MB
    const size_t NEEDED  = XB_OFF + XB_BYT + WB_BYT;              // 49 MB

    dim3 grid(Hsz / BN, Bsz / BM);
    if (ws_size >= NEEDED) {
        __bf16* xb  = (__bf16*)((char*)d_ws + XB_OFF);
        __bf16* wbT = (__bf16*)((char*)d_ws + XB_OFF + XB_BYT);
        convert_x_kernel<<<(Bsz * Dsz) / (256 * 4), 256, 0, stream>>>(x, xb);
        dim3 tgrid(Hsz / 32, Dsz / 32, Esz);
        transpose_w_kernel<<<tgrid, 256, 0, stream>>>(expert_w, wbT);
        moe_gemm_async<<<grid, 256, 0, stream>>>(xb, wbT, expert_b, gates, out);
    } else {
        moe_gemm_fallback<<<grid, 256, 0, stream>>>(x, expert_w, expert_b,
                                                    gates, out);
    }
}
